// SmyrfAttention_64158221467907
// MI455X (gfx1250) — matrix operand order repaired and checked by reading, not compiled, not run on hardware
//
#include <hip/hip_runtime.h>
#include <hip/hip_bf16.h>
#include <math.h>

// ---------------------------------------------------------------------------
// SMYRF attention forward for MI455X (gfx1250, wave32, WMMA, async-to-LDS).
// ---------------------------------------------------------------------------

typedef __attribute__((ext_vector_type(16))) __bf16 v16bf;
typedef __attribute__((ext_vector_type(8)))  __bf16 bf8;
typedef __attribute__((ext_vector_type(2)))  __bf16 bf2;
typedef __attribute__((ext_vector_type(8)))  float  v8f;
typedef int v4i __attribute__((vector_size(16)));   // matches async-LDS builtin proto

#define BS   32      // batch
#define SEQ  4097    // tokens incl. cls
#define QSQ  4096    // tokens excl. cls
#define DIM  64      // head dim
#define NH   4       // hashes
#define CH   128     // chunk size
#define NC   32      // chunks (4096/128)
#define MPAD 144     // 129 rows padded to 9*16
#define KPAD 160     // K-dim padding for P@V (5 steps of 32)
#define NWAVE 9
#define TPB  (NWAVE * 32)   // 288 threads = 9 waves (wave32)

// LDS carve (bytes) for attn_chunks
#define QS_OFF 0
#define KS_OFF 18432                 // 144*64*2
#define VS_OFF 36864                 // + 144*64*2
#define PS_OFF 57344                 // + 160*64*2 (V stored transposed [DIM][KPAD])
#define SMEM_TOTAL 103424            // + 9*16*160*2

// CDNA5 async global->LDS gather path (guarded; falls back to sync staging)
#if defined(__has_builtin)
#if __has_builtin(__builtin_amdgcn_global_load_async_to_lds_b128)
#define USE_ASYNC_GATHER 1
#endif
#endif
#ifndef USE_ASYNC_GATHER
#define USE_ASYNC_GATHER 0
#endif

__device__ __forceinline__ v8f zero_v8f() {
  v8f z;
#pragma unroll
  for (int i = 0; i < 8; ++i) z[i] = 0.f;
  return z;
}

// A-matrix 16x32 bf16 VGPR-pair K base (ISA 7.12.2): VGPR p<4 -> K=2p(+1) [+8 hi-half],
// p>=4 -> K=16+2(p-4)(+1) [+8 hi-half]
__device__ __forceinline__ int kbaseA(int p, int hf) {
  return ((p < 4) ? (2 * p) : (16 + 2 * (p - 4))) + hf * 8;
}

// Row reductions across a 16-lane half-wave (WMMA C layout: one row per half-wave)
__device__ __forceinline__ float redmax16(float x) {
  x = fmaxf(x, __int_as_float(__builtin_amdgcn_ds_swizzle(__float_as_int(x), 0x041f)));
  x = fmaxf(x, __int_as_float(__builtin_amdgcn_ds_swizzle(__float_as_int(x), 0x081f)));
  x = fmaxf(x, __int_as_float(__builtin_amdgcn_ds_swizzle(__float_as_int(x), 0x101f)));
  x = fmaxf(x, __int_as_float(__builtin_amdgcn_ds_swizzle(__float_as_int(x), 0x201f)));
  return x;
}
__device__ __forceinline__ float redsum16(float x) {
  x += __int_as_float(__builtin_amdgcn_ds_swizzle(__float_as_int(x), 0x041f));
  x += __int_as_float(__builtin_amdgcn_ds_swizzle(__float_as_int(x), 0x081f));
  x += __int_as_float(__builtin_amdgcn_ds_swizzle(__float_as_int(x), 0x101f));
  x += __int_as_float(__builtin_amdgcn_ds_swizzle(__float_as_int(x), 0x201f));
  return x;
}

// ---------------------------------------------------------------------------
// RNG constants (deterministic stand-in for jax.random key(1234) draws)
// ---------------------------------------------------------------------------
__global__ void init_rng(float* alpha, float* beta) {
  if (threadIdx.x != 0 || blockIdx.x != 0) return;
  unsigned s = 0x12345u ^ 1234u;
  for (int i = 0; i < (DIM + 2) * NH; i += 2) {
    s = s * 1664525u + 1013904223u;
    float u1 = (float)(s >> 8) * (1.0f / 16777216.0f) + 1e-7f;
    s = s * 1664525u + 1013904223u;
    float u2 = (float)(s >> 8) * (1.0f / 16777216.0f);
    float rr = sqrtf(-2.0f * __logf(u1));
    alpha[i]     = rr * __cosf(6.28318530718f * u2);
    alpha[i + 1] = rr * __sinf(6.28318530718f * u2);
  }
  for (int h = 0; h < NH; ++h) {
    s = s * 1664525u + 1013904223u;
    beta[h] = (float)(s >> 8) * (1.0f / 16777216.0f);  // * R (R==1)
  }
}

// ---------------------------------------------------------------------------
// One-pass fp32 -> bf16 copies of q,k,v (halves the later gather traffic and
// feeds WMMA directly). Streaming: 100 MB read + 50 MB write @ 23.3 TB/s.
// ---------------------------------------------------------------------------
__global__ __launch_bounds__(256) void cvt_bf16(const float* __restrict__ q,
                                                const float* __restrict__ k,
                                                const float* __restrict__ v,
                                                __bf16* __restrict__ qb,
                                                __bf16* __restrict__ kb,
                                                __bf16* __restrict__ vb) {
  size_t t = (size_t)blockIdx.x * blockDim.x + threadIdx.x;
  if (t >= (size_t)BS * SEQ * DIM) return;
  qb[t] = (__bf16)q[t];
  kb[t] = (__bf16)k[t];
  vb[t] = (__bf16)v[t];
}

// ---------------------------------------------------------------------------
// Row norms + per-batch M = max||q|| + max||k||
// ---------------------------------------------------------------------------
__global__ __launch_bounds__(256) void norms_kernel(const float* __restrict__ q,
                                                    const float* __restrict__ k,
                                                    float* __restrict__ qn,
                                                    float* __restrict__ kn,
                                                    float* __restrict__ Mout) {
  __shared__ float smq[256], smk[256];
  const int b = blockIdx.x, tid = threadIdx.x;
  const size_t base = (size_t)b * SEQ * DIM;
  float mq = -1e30f, mk = -1e30f;
  for (int i = tid; i < QSQ; i += 256) {
    const float* qr = q + base + (size_t)(i + 1) * DIM;
    const float* kr = k + base + (size_t)(i + 1) * DIM;
    float sq = 0.f, sk = 0.f;
#pragma unroll 16
    for (int d = 0; d < DIM; ++d) { sq += qr[d] * qr[d]; sk += kr[d] * kr[d]; }
    float nq = sqrtf(sq), nk = sqrtf(sk);
    qn[(size_t)b * QSQ + i] = nq;
    kn[(size_t)b * QSQ + i] = nk;
    mq = fmaxf(mq, nq); mk = fmaxf(mk, nk);
  }
  smq[tid] = mq; smk[tid] = mk;
  __syncthreads();
  for (int s = 128; s > 0; s >>= 1) {
    if (tid < s) {
      smq[tid] = fmaxf(smq[tid], smq[tid + s]);
      smk[tid] = fmaxf(smk[tid], smk[tid + s]);
    }
    __syncthreads();
  }
  if (tid == 0) Mout[b] = smq[0] + smk[0];
}

// ---------------------------------------------------------------------------
// Hash projections: qh = [q, ext_q, 0] @ alpha + beta ; kh = [k, 0, ext_k] @ alpha + beta
// ---------------------------------------------------------------------------
__global__ __launch_bounds__(256) void hash_kernel(const float* __restrict__ q,
                                                   const float* __restrict__ k,
                                                   const float* __restrict__ qn,
                                                   const float* __restrict__ kn,
                                                   const float* __restrict__ M,
                                                   const float* __restrict__ alpha,
                                                   const float* __restrict__ beta,
                                                   float* __restrict__ qhash,
                                                   float* __restrict__ khash) {
  int t = blockIdx.x * blockDim.x + threadIdx.x;
  if (t >= BS * QSQ) return;
  int b = t / QSQ, i = t - b * QSQ;
  float Mb = M[b];
  float nq = qn[t], nk = kn[t];
  float extq = sqrtf(fmaxf(Mb * Mb - nq * nq, 0.f));
  float extk = sqrtf(fmaxf(Mb * Mb - nk * nk, 0.f));
  float aq[NH], ak[NH];
#pragma unroll
  for (int h = 0; h < NH; ++h) {
    aq[h] = beta[h] + extq * alpha[64 * NH + h];
    ak[h] = beta[h] + extk * alpha[65 * NH + h];
  }
  const float* qr = q + ((size_t)b * SEQ + 1 + i) * DIM;
  const float* kr = k + ((size_t)b * SEQ + 1 + i) * DIM;
  for (int d = 0; d < DIM; ++d) {
    float qd = qr[d], kd = kr[d];
#pragma unroll
    for (int h = 0; h < NH; ++h) {
      aq[h] += qd * alpha[d * NH + h];
      ak[h] += kd * alpha[d * NH + h];
    }
  }
#pragma unroll
  for (int h = 0; h < NH; ++h) {
    qhash[((size_t)(h * BS + b)) * QSQ + i] = aq[h];
    khash[((size_t)(h * BS + b)) * QSQ + i] = ak[h];
  }
}

// ---------------------------------------------------------------------------
// In-LDS bitonic argsort of 4096 hash values per (hash,batch); y=0 -> q, y=1 -> k
// ---------------------------------------------------------------------------
__global__ __launch_bounds__(512) void sort_kernel(const float* __restrict__ qhash,
                                                   const float* __restrict__ khash,
                                                   int* __restrict__ q_pos,
                                                   int* __restrict__ k_pos) {
  __shared__ float key[QSQ];
  __shared__ int   idx[QSQ];
  const int hb = blockIdx.x;
  const float* src = (blockIdx.y == 0) ? qhash : khash;
  int* dst = (blockIdx.y == 0) ? q_pos : k_pos;
  const int tid = threadIdx.x;
  for (int i = tid; i < QSQ; i += 512) {
    key[i] = src[(size_t)hb * QSQ + i];
    idx[i] = i;
  }
  __syncthreads();
  for (int k2 = 2; k2 <= QSQ; k2 <<= 1) {
    for (int j = k2 >> 1; j > 0; j >>= 1) {
      for (int i = tid; i < QSQ; i += 512) {
        int ixj = i ^ j;
        if (ixj > i) {
          bool up = ((i & k2) == 0);
          float a = key[i], bv = key[ixj];
          if (up ? (a > bv) : (a < bv)) {
            key[i] = bv; key[ixj] = a;
            int ti = idx[i]; idx[i] = idx[ixj]; idx[ixj] = ti;
          }
        }
      }
      __syncthreads();
    }
  }
  for (int i = tid; i < QSQ; i += 512) dst[(size_t)hb * QSQ + i] = idx[i];
}

// ---------------------------------------------------------------------------
// Block attention: one workgroup per (chunk, batch, hash); 9 waves, one 16-row
// M-tile each. S = Qc*Kc^T via v_wmma_f32_16x16x32_bf16, fused softmax in the
// WMMA C layout, O = P*Vc via WMMA. Results scattered by q_pos.
// Q/K rows are gathered straight into LDS with GLOBAL_LOAD_ASYNC_TO_LDS_B128
// (ASYNCcnt) when the toolchain exposes the builtin.
// ---------------------------------------------------------------------------
__global__ __launch_bounds__(TPB) void attn_chunks(const __bf16* __restrict__ qb,
                                                   const __bf16* __restrict__ kb,
                                                   const __bf16* __restrict__ vb,
                                                   const int* __restrict__ qpos,
                                                   const int* __restrict__ kpos,
                                                   float* __restrict__ oTmp,
                                                   float* __restrict__ lseTmp,
                                                   float* __restrict__ clsTmp) {
  extern __shared__ char smem[];
  __bf16* Qs  = (__bf16*)(smem + QS_OFF);   // [MPAD][DIM] row-major
  __bf16* Ks  = (__bf16*)(smem + KS_OFF);   // [MPAD][DIM] row-major
  __bf16* VsT = (__bf16*)(smem + VS_OFF);   // [DIM][KPAD] (transposed)
  __bf16* Ps  = (__bf16*)(smem + PS_OFF);   // [NWAVE*16][KPAD]

  const int c = blockIdx.x, b = blockIdx.y, h = blockIdx.z;
  const int hb = h * BS + b;
  const int tid = threadIdx.x;
  const size_t gbase = (size_t)b * SEQ * DIM;
  const size_t pbase = (size_t)hb * QSQ + (size_t)c * CH;

#if USE_ASYNC_GATHER
  // ---- async gather of Q, K rows (129 rows x 128 B each) into LDS ----
  for (int u = tid; u < 129 * 8 * 2; u += TPB) {
    const int isk = (u >= 129 * 8);
    const int uu = isk ? (u - 129 * 8) : u;
    const int row = uu >> 3, cg = uu & 7;   // 8 x 16-byte groups per row
    int src = 0;
    if (row != 0) src = (isk ? kpos : qpos)[pbase + row - 1] + 1;
    const __bf16* g = (isk ? kb : qb) + gbase + (size_t)src * DIM + cg * 8;
    __bf16* l = (isk ? Ks : Qs) + (size_t)row * DIM + cg * 8;
    __builtin_amdgcn_global_load_async_to_lds_b128(
        (__attribute__((address_space(1))) v4i*)g,
        (__attribute__((address_space(3))) v4i*)l, 0, 0);
  }
  // zero the padded rows 129..143
  for (int e = 129 * DIM + tid; e < MPAD * DIM; e += TPB) {
    Qs[e] = (__bf16)0.f;
    Ks[e] = (__bf16)0.f;
  }
#else
  // ---- synchronous fallback staging of Q, K ----
  for (int e = tid; e < MPAD * DIM; e += TPB) {
    int row = e >> 6, col = e & 63;
    __bf16 qv = (__bf16)0.f, kv = (__bf16)0.f;
    if (row == 0) {
      qv = qb[gbase + col];
      kv = kb[gbase + col];
    } else if (row <= CH) {
      int jq = qpos[pbase + row - 1] + 1;
      int jk = kpos[pbase + row - 1] + 1;
      qv = qb[gbase + (size_t)jq * DIM + col];
      kv = kb[gbase + (size_t)jk * DIM + col];
    }
    Qs[e] = qv;
    Ks[e] = kv;
  }
#endif

  // ---- stage V transposed: VsT[col][row] (b128 row chunks, scalar DS scatter) ----
  for (int u = tid; u < KPAD * 8; u += TPB) {
    int row = u >> 3, cg = u & 7;
    bf8 vals;
    if (row == 0) {
      vals = *(const bf8*)(vb + gbase + cg * 8);
    } else if (row <= CH) {
      int jk = kpos[pbase + row - 1] + 1;
      vals = *(const bf8*)(vb + gbase + (size_t)jk * DIM + cg * 8);
    } else {
#pragma unroll
      for (int j = 0; j < 8; ++j) vals[j] = (__bf16)0.f;
    }
#pragma unroll
    for (int j = 0; j < 8; ++j) VsT[(size_t)(cg * 8 + j) * KPAD + row] = vals[j];
  }

#if USE_ASYNC_GATHER
#if defined(__has_builtin) && __has_builtin(__builtin_amdgcn_s_wait_asynccnt)
  __builtin_amdgcn_s_wait_asynccnt(0);
#else
  asm volatile("s_wait_asynccnt 0x0" ::: "memory");
#endif
#endif
  __syncthreads();

  const int w = tid >> 5, lane = tid & 31;
  const int m = lane & 15, hf = lane >> 4;

  // ---- matmul 1: S strip (16 x 144) ----
  v8f acc[9];
#pragma unroll
  for (int t = 0; t < 9; ++t) acc[t] = zero_v8f();

#pragma unroll
  for (int kk = 0; kk < DIM; kk += 32) {
    union { v16bf vv; bf2 p[8]; } A;
    const __bf16* qrow = Qs + (size_t)(w * 16 + m) * DIM + kk;
#pragma unroll
    for (int p = 0; p < 8; ++p) A.p[p] = *(const bf2*)(qrow + hf * 16 + 2 * p);
#pragma unroll
    for (int t = 0; t < 9; ++t) {
      v16bf B = *(const v16bf*)(Ks + (size_t)(t * 16 + m) * DIM + kk + hf * 16);
      acc[t] = __builtin_amdgcn_wmma_f32_16x16x32_bf16(false, A.vv, false, B,
                                                       (short)0, acc[t], false, false);
    }
  }

  // mask padded columns (valid cols 0..128; tile 8 keeps only col 128)
  if (m != 0) {
#pragma unroll
    for (int r = 0; r < 8; ++r) acc[8][r] = -1e30f;
  }

  // ---- fused softmax (rows live per 16-lane half-wave) ----
  float lse_r[8], inv_r[8];
#pragma unroll
  for (int r = 0; r < 8; ++r) {
    float mx = -1e30f;
#pragma unroll
    for (int t = 0; t < 9; ++t) mx = fmaxf(mx, acc[t][r]);
    mx = redmax16(mx);
    float s = 0.f;
#pragma unroll
    for (int t = 0; t < 9; ++t) {
      float p = __expf(acc[t][r] - mx);
      acc[t][r] = p;
      s += p;
    }
    s = redsum16(s);
    lse_r[r] = mx + __logf(s);
    inv_r[r] = 1.0f / s;
  }

  // ---- write normalized P (bf16) to LDS, zero K-pad cols [144,160) ----
#pragma unroll
  for (int r = 0; r < 8; ++r) {
    int prow = w * 16 + r + hf * 8;
#pragma unroll
    for (int t = 0; t < 9; ++t)
      Ps[(size_t)prow * KPAD + t * 16 + m] = (__bf16)(acc[t][r] * inv_r[r]);
    Ps[(size_t)prow * KPAD + MPAD + m] = (__bf16)0.f;
  }
  __syncthreads();

  // ---- matmul 2: O strip (16 x 64) = P (16 x 160) * V (160 x 64) ----
  v8f oacc[4];
#pragma unroll
  for (int vt = 0; vt < 4; ++vt) oacc[vt] = zero_v8f();

#pragma unroll
  for (int kk = 0; kk < KPAD; kk += 32) {
    union { v16bf vv; bf2 p[8]; } A;
    const __bf16* prow = Ps + (size_t)(w * 16 + m) * KPAD + kk;
#pragma unroll
    for (int p = 0; p < 8; ++p) A.p[p] = *(const bf2*)(prow + hf * 16 + 2 * p);
#pragma unroll
    for (int vt = 0; vt < 4; ++vt) {
      v16bf B = *(const v16bf*)(VsT + (size_t)(vt * 16 + m) * KPAD + kk + hf * 16);
      oacc[vt] = __builtin_amdgcn_wmma_f32_16x16x32_bf16(false, A.vv, false, B,
                                                         (short)0, oacc[vt], false, false);
    }
  }

  // ---- scatter outputs by q_pos; row 0 is the cls contribution ----
#pragma unroll
  for (int r = 0; r < 8; ++r) {
    int prow = w * 16 + r + hf * 8;
    if (prow == 0) {
      float* cp = clsTmp + ((size_t)hb * NC + c) * DIM;
#pragma unroll
      for (int vt = 0; vt < 4; ++vt) cp[vt * 16 + m] = oacc[vt][r];
    } else if (prow <= CH) {
      int j = qpos[pbase + prow - 1];
      float* op = oTmp + ((size_t)hb * QSQ + j) * DIM;
#pragma unroll
      for (int vt = 0; vt < 4; ++vt) op[vt * 16 + m] = oacc[vt][r];
      if (m == 0) lseTmp[(size_t)hb * QSQ + j] = lse_r[r];
    }
  }
}

// ---------------------------------------------------------------------------
// Final combine: softmax over the 4 hash logits, weighted sum of per-hash outs
// ---------------------------------------------------------------------------
__global__ __launch_bounds__(256) void combine_kernel(const float* __restrict__ oTmp,
                                                      const float* __restrict__ lseTmp,
                                                      float* __restrict__ out) {
  size_t t = (size_t)blockIdx.x * blockDim.x + threadIdx.x;
  if (t >= (size_t)BS * QSQ * DIM) return;
  int col = (int)(t & 63);
  size_t rj = t >> 6;
  int j = (int)(rj & (QSQ - 1));
  int b = (int)(rj >> 12);
  float lg[NH];
  float mx = -1e30f;
#pragma unroll
  for (int h = 0; h < NH; ++h) {
    lg[h] = lseTmp[((size_t)(h * BS + b)) * QSQ + j];
    mx = fmaxf(mx, lg[h]);
  }
  float s = 0.f, o = 0.f;
#pragma unroll
  for (int h = 0; h < NH; ++h) {
    float p = __expf(lg[h] - mx);
    s += p;
    o += p * oTmp[(((size_t)(h * BS + b)) * QSQ + j) * DIM + col];
  }
  out[((size_t)b * SEQ + 1 + j) * DIM + col] = o / s;
}

// cls row: mean over (hash, chunk) of the cls-block outputs
__global__ __launch_bounds__(64) void cls_kernel(const float* __restrict__ clsTmp,
                                                 float* __restrict__ out) {
  int b = blockIdx.x, col = threadIdx.x;
  float s = 0.f;
  for (int h = 0; h < NH; ++h)
    for (int c = 0; c < NC; ++c)
      s += clsTmp[(((size_t)(h * BS + b)) * NC + c) * DIM + col];
  out[(size_t)b * SEQ * DIM + col] = s * (1.0f / (NH * NC));
}

// ---------------------------------------------------------------------------
extern "C" void kernel_launch(void* const* d_in, const int* in_sizes, int n_in,
                              void* d_out, int out_size, void* d_ws, size_t ws_size,
                              hipStream_t stream) {
  (void)in_sizes; (void)n_in; (void)out_size; (void)ws_size;
  const float* q = (const float*)d_in[0];
  const float* k = (const float*)d_in[1];
  const float* v = (const float*)d_in[2];
  float* out = (float*)d_out;

  char* ws = (char*)d_ws;
  size_t off = 0;
  auto take = [&](size_t bytes) -> char* {
    char* p = ws + off;
    off += (bytes + 255) & ~(size_t)255;
    return p;
  };
  float*  alpha = (float*) take((size_t)(DIM + 2) * NH * 4);
  float*  beta  = (float*) take((size_t)NH * 4);
  float*  M     = (float*) take((size_t)BS * 4);
  float*  qn    = (float*) take((size_t)BS * QSQ * 4);
  float*  kn    = (float*) take((size_t)BS * QSQ * 4);
  float*  qhash = (float*) take((size_t)NH * BS * QSQ * 4);
  float*  khash = (float*) take((size_t)NH * BS * QSQ * 4);
  int*    qpos  = (int*)   take((size_t)NH * BS * QSQ * 4);
  int*    kpos  = (int*)   take((size_t)NH * BS * QSQ * 4);
  float*  lseT  = (float*) take((size_t)NH * BS * QSQ * 4);
  float*  clsT  = (float*) take((size_t)NH * BS * NC * DIM * 4);
  __bf16* qb    = (__bf16*)take((size_t)BS * SEQ * DIM * 2);
  __bf16* kb    = (__bf16*)take((size_t)BS * SEQ * DIM * 2);
  __bf16* vb    = (__bf16*)take((size_t)BS * SEQ * DIM * 2);
  float*  oT    = (float*) take((size_t)NH * BS * QSQ * DIM * 4);

  (void)hipFuncSetAttribute((const void*)attn_chunks,
                            hipFuncAttributeMaxDynamicSharedMemorySize, SMEM_TOTAL);

  init_rng<<<1, 64, 0, stream>>>(alpha, beta);
  cvt_bf16<<<(int)(((size_t)BS * SEQ * DIM + 255) / 256), 256, 0, stream>>>(q, k, v, qb, kb, vb);
  norms_kernel<<<BS, 256, 0, stream>>>(q, k, qn, kn, M);
  hash_kernel<<<(BS * QSQ + 255) / 256, 256, 0, stream>>>(q, k, qn, kn, M, alpha, beta,
                                                          qhash, khash);
  sort_kernel<<<dim3(NH * BS, 2), 512, 0, stream>>>(qhash, khash, qpos, kpos);
  attn_chunks<<<dim3(NC, BS, NH), TPB, SMEM_TOTAL, stream>>>(qb, kb, vb, qpos, kpos,
                                                             oT, lseT, clsT);
  combine_kernel<<<(int)(((size_t)BS * QSQ * DIM + 255) / 256), 256, 0, stream>>>(oT, lseT, out);
  cls_kernel<<<BS, DIM, 0, stream>>>(clsT, out);
}
